// Attention_qkv_Bias_65412351918528
// MI455X (gfx1250) — matrix operand order repaired and checked by reading, not compiled, not run on hardware
//
#include <hip/hip_runtime.h>
#include <hip/hip_bf16.h>

typedef __attribute__((ext_vector_type(16))) __bf16 bf16x16;
typedef __attribute__((ext_vector_type(8)))  float  f32x8;

#define Bn 4
#define Nseq 2048
#define Cc 768
#define Hn 12
#define HD 64
#define SCALE 0.125f   // 64^-0.5

// ---------------- helpers ----------------

__device__ __forceinline__ f32x8 wmma_bf16(bf16x16 a, bf16x16 b, f32x8 c) {
    return __builtin_amdgcn_wmma_f32_16x16x32_bf16(false, a, false, b, (short)0, c, false, false);
}

__device__ __forceinline__ bf16x16 load16(const __bf16* p) {
    bf16x16 v;
#pragma unroll
    for (int j = 0; j < 16; ++j) v[j] = p[j];
    return v;
}

// A-fragment (16x32 row-major, leading dim ld)
__device__ __forceinline__ bf16x16 load_a(const __bf16* base, int ld, int m0, int k0, int lane) {
    int m = lane & 15, hh = lane >> 4;
    const __bf16* p = base + (size_t)(m0 + m) * ld + k0 + 16 * hh;
    bf16x16 a;
#pragma unroll
    for (int j = 0; j < 8; ++j) { a[j] = p[j]; a[8 + j] = p[8 + j]; }
    return a;
}

// async global -> LDS, 16 bytes per lane (tracked by ASYNCcnt)
__device__ __forceinline__ void async_copy_b128(const __bf16* g, __bf16* lds_ptr) {
    unsigned lds = (unsigned)(size_t)lds_ptr;
    asm volatile("global_load_async_to_lds_b128 %0, %1, off" :: "v"(lds), "v"(g) : "memory");
}
template <int N>
__device__ __forceinline__ void async_wait_le() {
    asm volatile("s_wait_asynccnt %0" :: "n"(N) : "memory");
}

// ---------------- stage 0: conversions ----------------

__global__ void convert_bf16_kernel(const float* __restrict__ src, __bf16* __restrict__ dst, int n) {
    int i = blockIdx.x * blockDim.x + threadIdx.x;
    int stride = gridDim.x * blockDim.x;
    for (; i < n; i += stride) dst[i] = (__bf16)src[i];
}

// src [rows=K][cols=N] f32 -> dst [N][K] bf16 (pre-transposed weights)
__global__ void convert_transpose_kernel(const float* __restrict__ src, __bf16* __restrict__ dst,
                                         int rows, int cols) {
    int i = blockIdx.x * blockDim.x + threadIdx.x;
    int stride = gridDim.x * blockDim.x;
    int n = rows * cols;
    for (; i < n; i += stride) {
        int k = i / cols, c = i - k * cols;
        dst[(size_t)c * rows + k] = (__bf16)src[i];
    }
}

// ---------------- stage 1: QKV GEMM + bias scatter ----------------
// wt layout: [3C][C] transposed. grid (3C/64, B*N/64), 128 threads (4 waves)
__global__ __launch_bounds__(128) void gemm_qkv_kernel(
    const __bf16* __restrict__ xbf, const __bf16* __restrict__ wt,
    const float* __restrict__ b_qkv,
    const float* __restrict__ qbias, const float* __restrict__ kbias, const float* __restrict__ vbias,
    __bf16* __restrict__ qws, __bf16* __restrict__ kws, __bf16* __restrict__ vt_out) {

    __shared__ __bf16 wtile[2][64 * 32];       // ping-pong [n][kk]
    int tid = threadIdx.x, lane = tid & 31;
    int n = lane & 15, hh = lane >> 4;
    int nb = blockIdx.x * 64;
    int mb = blockIdx.y * 64 + (tid >> 5) * 16;

    int nn = tid >> 1, c4h = (tid & 1) * 2;    // 128 x (2x16B) chunks per tile
    const __bf16* wrow = wt + (size_t)(nb + nn) * Cc + c4h * 8;

    // prologue: tile 0
    async_copy_b128(wrow + 0, &wtile[0][nn * 32 + c4h * 8]);
    async_copy_b128(wrow + 8, &wtile[0][nn * 32 + c4h * 8 + 8]);

    f32x8 acc[4] = {};
    const int T = Cc / 32;                     // 24
    for (int i = 0; i < T; ++i) {
        int k0 = i * 32, cur = i & 1;
        if (i + 1 < T) {
            const __bf16* g = wrow + (k0 + 32);
            async_copy_b128(g + 0, &wtile[cur ^ 1][nn * 32 + c4h * 8]);
            async_copy_b128(g + 8, &wtile[cur ^ 1][nn * 32 + c4h * 8 + 8]);
            async_wait_le<2>();                // tile i done; tile i+1 in flight
        } else {
            async_wait_le<0>();
        }
        __syncthreads();
        bf16x16 a  = load_a(xbf, Cc, mb, k0, lane);
        bf16x16 b0 = load16(&wtile[cur][(0 * 16 + n) * 32 + 16 * hh]);
        bf16x16 b1 = load16(&wtile[cur][(1 * 16 + n) * 32 + 16 * hh]);
        bf16x16 b2 = load16(&wtile[cur][(2 * 16 + n) * 32 + 16 * hh]);
        bf16x16 b3 = load16(&wtile[cur][(3 * 16 + n) * 32 + 16 * hh]);
        acc[0] = wmma_bf16(a, b0, acc[0]);
        acc[1] = wmma_bf16(a, b1, acc[1]);
        acc[2] = wmma_bf16(a, b2, acc[2]);
        acc[3] = wmma_bf16(a, b3, acc[3]);
        __syncthreads();
    }

    // uniform per block: 64-wide tile never straddles q/k/v or head boundaries
    int s  = nb / Cc;                          // 0=q 1=k 2=v (scalar)
    int hd = (nb - s * Cc) >> 6;               // head (scalar)
#pragma unroll
    for (int dn = 0; dn < 4; ++dn) {
#pragma unroll
        for (int r = 0; r < 8; ++r) {
            int row = mb + r + 8 * hh;
            int d   = dn * 16 + n;
            float val = acc[dn][r] + b_qkv[nb + d];
            int bb = row >> 11, pos = row & (Nseq - 1);
            size_t std_i = (((size_t)(bb * Hn + hd)) * Nseq + pos) * HD + d;
            if (s == 0) {
                qws[std_i] = (__bf16)(val + qbias[std_i]);
            } else if (s == 1) {
                kws[std_i] = (__bf16)(val + kbias[std_i]);
            } else {
                size_t vt_i = (((size_t)(bb * Hn + hd)) * HD + d) * Nseq + pos;
                vt_out[vt_i] = (__bf16)(val + vbias[std_i]);
            }
        }
    }
}

// ---------------- stage 2: flash attention ----------------
// grid (N/64, B*H), 128 threads; wave w -> q rows [blockIdx.x*64 + 16w, +16)
__global__ __launch_bounds__(128) void attn_kernel(
    const __bf16* __restrict__ qws, const __bf16* __restrict__ kws,
    const __bf16* __restrict__ vt, __bf16* __restrict__ attn_out) {

    __shared__ __bf16 ktile[2][32 * 64];       // [pos][d]
    __shared__ __bf16 vtile[2][64 * 32];       // [d][pos]
    __shared__ __bf16 pbuf[4][16 * 32];

    int tid = threadIdx.x, lane = tid & 31, wave = tid >> 5;
    int n = lane & 15, hh = lane >> 4;
    int bh = blockIdx.y;
    int b = bh / Hn, h = bh % Hn;

    const __bf16* qb = qws + (size_t)bh * Nseq * HD;
    const __bf16* kb = kws + (size_t)bh * Nseq * HD;
    const __bf16* vb = vt  + (size_t)bh * HD * Nseq;

    int q0 = blockIdx.x * 64 + wave * 16;
    bf16x16 aq0 = load_a(qb, HD, q0, 0, lane);
    bf16x16 aq1 = load_a(qb, HD, q0, 32, lane);

    // staging addresses (per thread, fixed shapes)
    int kidx = tid;                            // K tile: 256 chunks of 16B, 2 per thread
    int vd = tid >> 1, vc = (tid & 1) * 2;     // V tile: row d, 2 adjacent 16B chunks

    // prologue: tile 0
    {
        async_copy_b128(kb + (size_t)kidx * 8,       &ktile[0][kidx * 8]);
        async_copy_b128(kb + (size_t)(kidx+128) * 8, &ktile[0][(kidx+128) * 8]);
        const __bf16* g = vb + (size_t)vd * Nseq + vc * 8;
        async_copy_b128(g + 0, &vtile[0][vd * 32 + vc * 8]);
        async_copy_b128(g + 8, &vtile[0][vd * 32 + vc * 8 + 8]);
    }

    f32x8 O[4] = {};
    float mrow[8], lrow[8];
#pragma unroll
    for (int r = 0; r < 8; ++r) { mrow[r] = -1e30f; lrow[r] = 0.0f; }

    const int T = Nseq / 32;                   // 64
    for (int i = 0; i < T; ++i) {
        int kp = i * 32, cur = i & 1;
        if (i + 1 < T) {
            const __bf16* gk = kb + (size_t)(kp + 32) * HD;
            async_copy_b128(gk + (size_t)kidx * 8,       &ktile[cur ^ 1][kidx * 8]);
            async_copy_b128(gk + (size_t)(kidx+128) * 8, &ktile[cur ^ 1][(kidx+128) * 8]);
            const __bf16* gv = vb + (size_t)vd * Nseq + (kp + 32) + vc * 8;
            async_copy_b128(gv + 0, &vtile[cur ^ 1][vd * 32 + vc * 8]);
            async_copy_b128(gv + 8, &vtile[cur ^ 1][vd * 32 + vc * 8 + 8]);
            async_wait_le<4>();                // tile i done; tile i+1 in flight
        } else {
            async_wait_le<0>();
        }
        __syncthreads();

        // S = q @ k^T  (16 x 32 keypos)
        bf16x16 bk00 = load16(&ktile[cur][n * 64 + 0  + 16 * hh]);
        bf16x16 bk01 = load16(&ktile[cur][n * 64 + 32 + 16 * hh]);
        bf16x16 bk10 = load16(&ktile[cur][(n + 16) * 64 + 0  + 16 * hh]);
        bf16x16 bk11 = load16(&ktile[cur][(n + 16) * 64 + 32 + 16 * hh]);
        f32x8 S0 = {}, S1 = {};
        S0 = wmma_bf16(aq0, bk00, S0);
        S0 = wmma_bf16(aq1, bk01, S0);
        S1 = wmma_bf16(aq0, bk10, S1);
        S1 = wmma_bf16(aq1, bk11, S1);

        // online softmax; row M=r+8*hh lives across 16-lane half-waves
#pragma unroll
        for (int r = 0; r < 8; ++r) {
            float s0 = S0[r] * SCALE, s1 = S1[r] * SCALE;
            float rmax = fmaxf(s0, s1);
#pragma unroll
            for (int off = 8; off >= 1; off >>= 1)
                rmax = fmaxf(rmax, __shfl_xor(rmax, off, 16));
            float mnew = fmaxf(mrow[r], rmax);
            float alpha = __expf(mrow[r] - mnew);
            float p0 = __expf(s0 - mnew);
            float p1 = __expf(s1 - mnew);
            float rsum = p0 + p1;
#pragma unroll
            for (int off = 8; off >= 1; off >>= 1)
                rsum += __shfl_xor(rsum, off, 16);
            lrow[r] = lrow[r] * alpha + rsum;
            mrow[r] = mnew;
#pragma unroll
            for (int dn = 0; dn < 4; ++dn) O[dn][r] *= alpha;
            pbuf[wave][(r + 8 * hh) * 32 + n]      = (__bf16)p0;
            pbuf[wave][(r + 8 * hh) * 32 + 16 + n] = (__bf16)p1;
        }
        asm volatile("s_wait_dscnt 0" ::: "memory");

        // reload P in A-layout (K-dim = 32 key positions)
        bf16x16 ap;
        {
            const __bf16* pp = &pbuf[wave][n * 32];
#pragma unroll
            for (int j = 0; j < 8; ++j) {
                ap[j]     = pp[16 * hh + j];
                ap[8 + j] = pp[16 * hh + 8 + j];
            }
        }
        // O += P @ V
        bf16x16 bv0 = load16(&vtile[cur][(0 * 16 + n) * 32 + 16 * hh]);
        bf16x16 bv1 = load16(&vtile[cur][(1 * 16 + n) * 32 + 16 * hh]);
        bf16x16 bv2 = load16(&vtile[cur][(2 * 16 + n) * 32 + 16 * hh]);
        bf16x16 bv3 = load16(&vtile[cur][(3 * 16 + n) * 32 + 16 * hh]);
        O[0] = wmma_bf16(ap, bv0, O[0]);
        O[1] = wmma_bf16(ap, bv1, O[1]);
        O[2] = wmma_bf16(ap, bv2, O[2]);
        O[3] = wmma_bf16(ap, bv3, O[3]);
        __syncthreads();
    }

    // normalize and scatter to [B*N, C] bf16 with col = h*64 + d
#pragma unroll
    for (int r = 0; r < 8; ++r) {
        float inv = 1.0f / lrow[r];
        int pos = q0 + r + 8 * hh;
        size_t row = (size_t)b * Nseq + pos;
#pragma unroll
        for (int dn = 0; dn < 4; ++dn)
            attn_out[row * Cc + h * HD + dn * 16 + n] = (__bf16)(O[dn][r] * inv);
    }
}

// ---------------- stage 3: output projection ----------------
__global__ __launch_bounds__(128) void gemm_proj_kernel(
    const __bf16* __restrict__ abf, const __bf16* __restrict__ wt,
    const float* __restrict__ b_proj, float* __restrict__ out) {

    __shared__ __bf16 wtile[2][64 * 32];
    int tid = threadIdx.x, lane = tid & 31;
    int n = lane & 15, hh = lane >> 4;
    int nb = blockIdx.x * 64;
    int mb = blockIdx.y * 64 + (tid >> 5) * 16;

    int nn = tid >> 1, c4h = (tid & 1) * 2;
    const __bf16* wrow = wt + (size_t)(nb + nn) * Cc + c4h * 8;

    async_copy_b128(wrow + 0, &wtile[0][nn * 32 + c4h * 8]);
    async_copy_b128(wrow + 8, &wtile[0][nn * 32 + c4h * 8 + 8]);

    f32x8 acc[4] = {};
    const int T = Cc / 32;
    for (int i = 0; i < T; ++i) {
        int k0 = i * 32, cur = i & 1;
        if (i + 1 < T) {
            const __bf16* g = wrow + (k0 + 32);
            async_copy_b128(g + 0, &wtile[cur ^ 1][nn * 32 + c4h * 8]);
            async_copy_b128(g + 8, &wtile[cur ^ 1][nn * 32 + c4h * 8 + 8]);
            async_wait_le<2>();
        } else {
            async_wait_le<0>();
        }
        __syncthreads();
        bf16x16 a  = load_a(abf, Cc, mb, k0, lane);
        bf16x16 b0 = load16(&wtile[cur][(0 * 16 + n) * 32 + 16 * hh]);
        bf16x16 b1 = load16(&wtile[cur][(1 * 16 + n) * 32 + 16 * hh]);
        bf16x16 b2 = load16(&wtile[cur][(2 * 16 + n) * 32 + 16 * hh]);
        bf16x16 b3 = load16(&wtile[cur][(3 * 16 + n) * 32 + 16 * hh]);
        acc[0] = wmma_bf16(a, b0, acc[0]);
        acc[1] = wmma_bf16(a, b1, acc[1]);
        acc[2] = wmma_bf16(a, b2, acc[2]);
        acc[3] = wmma_bf16(a, b3, acc[3]);
        __syncthreads();
    }
#pragma unroll
    for (int dn = 0; dn < 4; ++dn)
#pragma unroll
        for (int r = 0; r < 8; ++r) {
            int row = mb + r + 8 * hh;
            int col = nb + dn * 16 + n;
            out[(size_t)row * Cc + col] = acc[dn][r] + b_proj[col];
        }
}

// ---------------- host ----------------

extern "C" void kernel_launch(void* const* d_in, const int* in_sizes, int n_in,
                              void* d_out, int out_size, void* d_ws, size_t ws_size,
                              hipStream_t stream) {
    const float* x      = (const float*)d_in[0];
    const float* qbias  = (const float*)d_in[1];
    const float* kbias  = (const float*)d_in[2];
    const float* vbias  = (const float*)d_in[3];
    const float* W_qkv  = (const float*)d_in[4];
    const float* b_qkv  = (const float*)d_in[5];
    const float* W_proj = (const float*)d_in[6];
    const float* b_proj = (const float*)d_in[7];
    float* out = (float*)d_out;

    const size_t n_x     = (size_t)Bn * Nseq * Cc;
    const size_t n_wqkv  = (size_t)Cc * 3 * Cc;
    const size_t n_wproj = (size_t)Cc * Cc;
    const size_t n_qkv1  = (size_t)Bn * Hn * Nseq * HD;

    char* ws = (char*)d_ws;
    __bf16* xbf    = (__bf16*)ws;                  ws += n_x * 2;
    __bf16* wqkvT  = (__bf16*)ws;                  ws += n_wqkv * 2;   // [3C][C]
    __bf16* wprojT = (__bf16*)ws;                  ws += n_wproj * 2;  // [C][C]
    __bf16* qws    = (__bf16*)ws;                  ws += n_qkv1 * 2;   // [B,H,N,HD]
    __bf16* kws    = (__bf16*)ws;                  ws += n_qkv1 * 2;   // [B,H,N,HD]
    __bf16* vt     = (__bf16*)ws;                  ws += n_qkv1 * 2;   // [B,H,HD,N]
    __bf16* attnbf = (__bf16*)ws;                  ws += n_x * 2;      // [B*N,C]

    convert_bf16_kernel<<<1024, 256, 0, stream>>>(x, xbf, (int)n_x);
    convert_transpose_kernel<<<1024, 256, 0, stream>>>(W_qkv, wqkvT, Cc, 3 * Cc);
    convert_transpose_kernel<<<512, 256, 0, stream>>>(W_proj, wprojT, Cc, Cc);

    gemm_qkv_kernel<<<dim3(3 * Cc / 64, (Bn * Nseq) / 64), 128, 0, stream>>>(
        xbf, wqkvT, b_qkv, qbias, kbias, vbias, qws, kws, vt);

    attn_kernel<<<dim3(Nseq / 64, Bn * Hn), 128, 0, stream>>>(qws, kws, vt, attnbf);

    gemm_proj_kernel<<<dim3(Cc / 64, (Bn * Nseq) / 64), 128, 0, stream>>>(
        attnbf, wprojT, b_proj, out);

    (void)in_sizes; (void)n_in; (void)out_size; (void)ws_size;
}
